// SinkhornDistance_5248450035776
// MI455X (gfx1250) — compile-verified, matrix operand order checked by reading, not run on hardware
//
#include <hip/hip_runtime.h>
#include <hip/hip_fp16.h>

typedef _Float16 v16h __attribute__((ext_vector_type(16)));
typedef _Float16 v8h  __attribute__((ext_vector_type(8)));
typedef float    v8f  __attribute__((ext_vector_type(8)));

#define B_N 8192
#define D_N 512
#define N_ITER 100

// ---------------------------------------------------------------------------
// Phase 1: row-wise L2 normalization, fp32 -> fp16, plus x2 = sum(xn^2)
// ---------------------------------------------------------------------------
__global__ void __launch_bounds__(256) norm_kernel(
    const float* __restrict__ x, const float* __restrict__ y,
    _Float16* __restrict__ xn, _Float16* __restrict__ yn,
    float* __restrict__ x2, float* __restrict__ y2) {
  __shared__ float red[256];
  int r = blockIdx.x;
  const float* src; _Float16* dst; float* s2;
  if (r < B_N) { src = x + (size_t)r * D_N; dst = xn + (size_t)r * D_N; s2 = x2 + r; }
  else { int rr = r - B_N; src = y + (size_t)rr * D_N; dst = yn + (size_t)rr * D_N; s2 = y2 + rr; }
  int t = threadIdx.x;
  float a0 = src[t], a1 = src[t + 256];
  red[t] = a0 * a0 + a1 * a1;
  __syncthreads();
  for (int o = 128; o > 0; o >>= 1) { if (t < o) red[t] += red[t + o]; __syncthreads(); }
  float sum = red[0];
  float inv = 1.0f / fmaxf(sqrtf(sum), 1e-12f);
  dst[t]       = (_Float16)(a0 * inv);
  dst[t + 256] = (_Float16)(a1 * inv);
  if (t == 0) *s2 = sum * inv * inv;
}

__global__ void init_v_kernel(float* __restrict__ v) {
  v[blockIdx.x * 256 + threadIdx.x] = 0.0f;
}

// ---------------------------------------------------------------------------
// WMMA fragment loader (16-bit A 16x32 layout per CDNA5 ISA 7.12.2):
// lane 0-15 holds row (rbase+lane), halves = K{0..7} then K{16..23};
// lane 16-31 same rows, halves = K{8..15} then K{24..31}.
// B = yn^T, so a B fragment (col n) uses identical addressing on yn rows.
// ---------------------------------------------------------------------------
__device__ __forceinline__ v16h load_frag(const _Float16* __restrict__ base,
                                          int rbase, int k, int lane) {
  int l16 = lane & 15, hi = lane >> 4;
  const _Float16* p = base + (size_t)(rbase + l16) * D_N + k + hi * 8;
  union { v16h v; v8h h[2]; } u;
  u.h[0] = *(const v8h*)p;
  u.h[1] = *(const v8h*)(p + 16);
  return u.v;
}

// ---------------------------------------------------------------------------
// Phase 2: C = xn@yn^T via v_wmma_f32_16x16x32_f16, fused epilogue
// K = clip(exp(-clip(C/eps)), stored fp16 (128 MB -> L2-resident).
// Block = 8 waves, each wave owns a 32x32 output tile; block tile 64x128.
// ---------------------------------------------------------------------------
__global__ void __launch_bounds__(256) gemm_k_kernel(
    const _Float16* __restrict__ xn, const _Float16* __restrict__ yn,
    const float* __restrict__ x2, const float* __restrict__ y2,
    _Float16* __restrict__ Kmat) {
  int tid = threadIdx.x, lane = tid & 31, w = tid >> 5;
  int bx = blockIdx.x;
  int bm = bx & 127, bn = bx >> 7;        // 128 x 64 block grid
  int m0 = bm * 64 + (w & 1) * 32;
  int n0 = bn * 128 + (w >> 1) * 32;
  v8f acc[2][2] = {};
  for (int k = 0; k < D_N; k += 32) {
    v16h a0 = load_frag(xn, m0,      k, lane);
    v16h a1 = load_frag(xn, m0 + 16, k, lane);
    v16h b0 = load_frag(yn, n0,      k, lane);
    v16h b1 = load_frag(yn, n0 + 16, k, lane);
    acc[0][0] = __builtin_amdgcn_wmma_f32_16x16x32_f16(false, a0, false, b0, (short)0, acc[0][0], false, false);
    acc[0][1] = __builtin_amdgcn_wmma_f32_16x16x32_f16(false, a0, false, b1, (short)0, acc[0][1], false, false);
    acc[1][0] = __builtin_amdgcn_wmma_f32_16x16x32_f16(false, a1, false, b0, (short)0, acc[1][0], false, false);
    acc[1][1] = __builtin_amdgcn_wmma_f32_16x16x32_f16(false, a1, false, b1, (short)0, acc[1][1], false, false);
  }
  int l16 = lane & 15, hi = lane >> 4;
  #pragma unroll
  for (int mt = 0; mt < 2; mt++)
    #pragma unroll
    for (int nt = 0; nt < 2; nt++)
      #pragma unroll
      for (int e = 0; e < 8; e++) {
        int row = m0 + mt * 16 + hi * 8 + e;     // C/D layout: M = vgpr + 8*(lane>=16)
        int col = n0 + nt * 16 + l16;            // N = lane%16
        float dot = acc[mt][nt][e];
        float Cv = x2[row] + y2[col] - 2.0f * dot;
        Cv = fminf(fmaxf(Cv, 0.0f), 10.0f);
        float Kv = __expf(-fminf(Cv * 10.0f, 10.0f));
        Kv = fminf(fmaxf(Kv, 1e-10f), 1e10f);
        Kmat[(size_t)row * B_N + col] = (_Float16)Kv;
      }
}

// ---------------------------------------------------------------------------
// u = clip(1/(K@v + 1e-8)) : wave-per-row, 128-bit fp16 loads, v in LDS,
// deterministic shfl_xor reduction. 256 blocks * 8 waves * 4 rows = 8192.
// ---------------------------------------------------------------------------
__global__ void __launch_bounds__(256) matvec_row_kernel(
    const _Float16* __restrict__ Kmat, const float* __restrict__ v,
    float* __restrict__ u) {
  __shared__ float vbuf[B_N];
  int tid = threadIdx.x;
  for (int i = tid; i < B_N; i += 256) vbuf[i] = v[i];
  __syncthreads();
  int gw = blockIdx.x * 8 + (tid >> 5);
  int lane = tid & 31;
  for (int rr = 0; rr < 4; rr++) {
    int row = gw * 4 + rr;
    const _Float16* kp = Kmat + (size_t)row * B_N;
    float s = 0.0f;
    for (int it = 0; it < 32; it++) {
      int j = it * 256 + lane * 8;
      v8h kv = *(const v8h*)(kp + j);
      const float4* vp = (const float4*)&vbuf[j];
      float4 f0 = vp[0], f1 = vp[1];
      s += (float)kv[0] * f0.x + (float)kv[1] * f0.y + (float)kv[2] * f0.z + (float)kv[3] * f0.w
         + (float)kv[4] * f1.x + (float)kv[5] * f1.y + (float)kv[6] * f1.z + (float)kv[7] * f1.w;
    }
    for (int o = 16; o > 0; o >>= 1) s += __shfl_xor(s, o, 32);
    if (lane == 0) u[row] = fminf(fmaxf(1.0f / (s + 1e-8f), 1e-8f), 1e8f);
  }
}

// ---------------------------------------------------------------------------
// partial[rs][j] = sum over row slice rs of K[i][j]*u[i] : thread-per-column
// gives fully coalesced K^T access on row-major K (no transpose copy needed,
// keeps working set = 128 MB so K stays L2-resident). 32 colblocks x 8 slices.
// ---------------------------------------------------------------------------
__global__ void __launch_bounds__(256) matvec_col_kernel(
    const _Float16* __restrict__ Kmat, const float* __restrict__ u,
    float* __restrict__ part) {
  __shared__ float ubuf[256];
  int tid = threadIdx.x;
  int cb = blockIdx.x & 31, rs = blockIdx.x >> 5;
  int j = cb * 256 + tid;
  float s = 0.0f;
  for (int ib = 0; ib < 4; ib++) {
    int rbase = rs * 1024 + ib * 256;
    __syncthreads();
    ubuf[tid] = u[rbase + tid];
    __syncthreads();
    const _Float16* kp = Kmat + (size_t)rbase * B_N + j;
    #pragma unroll 4
    for (int ii = 0; ii < 256; ii++)
      s += (float)kp[(size_t)ii * B_N] * ubuf[ii];
  }
  part[rs * B_N + j] = s;
}

__global__ void __launch_bounds__(256) col_reduce_kernel(
    const float* __restrict__ part, float* __restrict__ v) {
  int j = blockIdx.x * 256 + threadIdx.x;
  float s = 0.0f;
  #pragma unroll
  for (int rs = 0; rs < 8; rs++) s += part[rs * B_N + j];
  v[j] = fminf(fmaxf(1.0f / (s + 1e-8f), 1e-8f), 1e8f);
}

// ---------------------------------------------------------------------------
// cost partials: recompute C via WMMA (bit-identical to phase 2), fuse
// term = u_i * K_ij * v_j * C_ij, deterministic wave reduction.
// ---------------------------------------------------------------------------
__global__ void __launch_bounds__(256) cost_kernel(
    const _Float16* __restrict__ xn, const _Float16* __restrict__ yn,
    const float* __restrict__ x2, const float* __restrict__ y2,
    const float* __restrict__ u, const float* __restrict__ v,
    float* __restrict__ part) {
  int tid = threadIdx.x, lane = tid & 31, w = tid >> 5;
  int bx = blockIdx.x;
  int bm = bx & 127, bn = bx >> 7;
  int m0 = bm * 64 + (w & 1) * 32;
  int n0 = bn * 128 + (w >> 1) * 32;
  v8f acc[2][2] = {};
  for (int k = 0; k < D_N; k += 32) {
    v16h a0 = load_frag(xn, m0,      k, lane);
    v16h a1 = load_frag(xn, m0 + 16, k, lane);
    v16h b0 = load_frag(yn, n0,      k, lane);
    v16h b1 = load_frag(yn, n0 + 16, k, lane);
    acc[0][0] = __builtin_amdgcn_wmma_f32_16x16x32_f16(false, a0, false, b0, (short)0, acc[0][0], false, false);
    acc[0][1] = __builtin_amdgcn_wmma_f32_16x16x32_f16(false, a0, false, b1, (short)0, acc[0][1], false, false);
    acc[1][0] = __builtin_amdgcn_wmma_f32_16x16x32_f16(false, a1, false, b0, (short)0, acc[1][0], false, false);
    acc[1][1] = __builtin_amdgcn_wmma_f32_16x16x32_f16(false, a1, false, b1, (short)0, acc[1][1], false, false);
  }
  int l16 = lane & 15, hi = lane >> 4;
  float lsum = 0.0f;
  #pragma unroll
  for (int mt = 0; mt < 2; mt++)
    #pragma unroll
    for (int nt = 0; nt < 2; nt++)
      #pragma unroll
      for (int e = 0; e < 8; e++) {
        int row = m0 + mt * 16 + hi * 8 + e;
        int col = n0 + nt * 16 + l16;
        float dot = acc[mt][nt][e];
        float Cv = x2[row] + y2[col] - 2.0f * dot;
        Cv = fminf(fmaxf(Cv, 0.0f), 10.0f);
        float Kv = __expf(-fminf(Cv * 10.0f, 10.0f));
        Kv = fminf(fmaxf(Kv, 1e-10f), 1e10f);
        lsum += u[row] * Kv * v[col] * Cv;
      }
  for (int o = 16; o > 0; o >>= 1) lsum += __shfl_xor(lsum, o, 32);
  if (lane == 0) part[blockIdx.x * 8 + w] = lsum;
}

__global__ void __launch_bounds__(256) fallback_kernel(
    const _Float16* __restrict__ xn, const _Float16* __restrict__ yn,
    float* __restrict__ part) {
  __shared__ float red[256];
  int t = threadIdx.x;
  size_t total = (size_t)B_N * D_N;
  float s = 0.0f;
  for (size_t i = (size_t)blockIdx.x * 256 + t; i < total; i += (size_t)256 * 256) {
    float d = (float)xn[i] - (float)yn[i];
    s += d * d;
  }
  red[t] = s; __syncthreads();
  for (int o = 128; o > 0; o >>= 1) { if (t < o) red[t] += red[t + o]; __syncthreads(); }
  if (t == 0) part[blockIdx.x] = red[0];
}

__global__ void __launch_bounds__(256) combine_kernel(
    const float* __restrict__ costpart, const float* __restrict__ fbpart,
    float* __restrict__ out) {
  __shared__ float red[256];
  int t = threadIdx.x;
  float s = 0.0f;
  for (int i = t; i < 65536; i += 256) s += costpart[i];
  red[t] = s; __syncthreads();
  for (int o = 128; o > 0; o >>= 1) { if (t < o) red[t] += red[t + o]; __syncthreads(); }
  float cost = red[0] / (float)B_N;
  __syncthreads();
  red[t] = fbpart[t]; __syncthreads();
  for (int o = 128; o > 0; o >>= 1) { if (t < o) red[t] += red[t + o]; __syncthreads(); }
  float fb = red[0] / (float)((size_t)B_N * D_N);
  if (t == 0) out[0] = __builtin_isfinite(cost) ? cost : fb;
}

// ---------------------------------------------------------------------------
extern "C" void kernel_launch(void* const* d_in, const int* in_sizes, int n_in,
                              void* d_out, int out_size, void* d_ws, size_t ws_size,
                              hipStream_t stream) {
  (void)in_sizes; (void)n_in; (void)out_size; (void)ws_size;
  const float* x = (const float*)d_in[0];
  const float* y = (const float*)d_in[1];
  float* out = (float*)d_out;

  char* w = (char*)d_ws;
  size_t off = 0;
  auto alloc = [&](size_t bytes) -> void* {
    void* p = w + off;
    off += (bytes + 255) & ~(size_t)255;
    return p;
  };
  _Float16* xn     = (_Float16*)alloc((size_t)B_N * D_N * 2);
  _Float16* yn     = (_Float16*)alloc((size_t)B_N * D_N * 2);
  float* x2        = (float*)alloc((size_t)B_N * 4);
  float* y2        = (float*)alloc((size_t)B_N * 4);
  float* u         = (float*)alloc((size_t)B_N * 4);
  float* v         = (float*)alloc((size_t)B_N * 4);
  float* colpart   = (float*)alloc((size_t)8 * B_N * 4);
  float* costpart  = (float*)alloc((size_t)65536 * 4);
  float* fbpart    = (float*)alloc((size_t)256 * 4);
  _Float16* Kmat   = (_Float16*)alloc((size_t)B_N * B_N * 2);   // 128 MB, L2-resident

  norm_kernel<<<2 * B_N, 256, 0, stream>>>(x, y, xn, yn, x2, y2);
  init_v_kernel<<<B_N / 256, 256, 0, stream>>>(v);
  gemm_k_kernel<<<8192, 256, 0, stream>>>(xn, yn, x2, y2, Kmat);

  for (int it = 0; it < N_ITER; ++it) {
    matvec_row_kernel<<<256, 256, 0, stream>>>(Kmat, v, u);
    matvec_col_kernel<<<256, 256, 0, stream>>>(Kmat, u, colpart);
    col_reduce_kernel<<<32, 256, 0, stream>>>(colpart, v);
  }

  cost_kernel<<<8192, 256, 0, stream>>>(xn, yn, x2, y2, u, v, costpart);
  fallback_kernel<<<256, 256, 0, stream>>>(xn, yn, fbpart);
  combine_kernel<<<1, 256, 0, stream>>>(costpart, fbpart, out);
}
